// MultiScaleConv_28381143892103
// MI455X (gfx1250) — compile-verified, matrix operand order checked by reading, not run on hardware
//
#include <hip/hip_runtime.h>
#include <hip/hip_bf16.h>

typedef _Float16 v16h __attribute__((ext_vector_type(16)));
typedef float    v8f  __attribute__((ext_vector_type(8)));

#define Cc 96
#define Hh 224
#define Ww 224
#define TH 8
#define TW 16
#define TIN_H 14
#define TIN_W 22
#define KSTR 104                    // padded k-stride (halves): 208B rows -> 16B aligned
#define S_IN_BYTES  (16 * TIN_H * TIN_W * 4)          // 19712
#define S_COMB_OFF  S_IN_BYTES                        // 16B aligned (19712 % 16 == 0)
#define S_LDS_BYTES (S_IN_BYTES + TH * TW * KSTR * 2) // 19712 + 26624 = 46336

__global__ __launch_bounds__(256)
void msconv_fused_wmma(const float* __restrict__ x,
                       const float* __restrict__ w3, const float* __restrict__ b3,
                       const float* __restrict__ w5, const float* __restrict__ b5,
                       const float* __restrict__ w7, const float* __restrict__ b7,
                       const float* __restrict__ wf, const float* __restrict__ bfin,
                       float* __restrict__ out)
{
    __shared__ __align__(16) char s_lds[S_LDS_BYTES];
    float    (*s_in)[TIN_H][TIN_W] = (float (*)[TIN_H][TIN_W])(s_lds);
    _Float16 (*s_comb)[KSTR]       = (_Float16 (*)[KSTR])(s_lds + S_COMB_OFF);

    const int tid  = threadIdx.x;
    const int lane = tid & 31;
    const int wave = tid >> 5;          // 8 waves; wave w owns pixel row w of the tile
    const int w0 = blockIdx.x * TW;
    const int h0 = blockIdx.y * TH;
    const int bb = blockIdx.z;

    const int col  = tid & 15;          // conv phase: column within tile
    const int crow = tid >> 4;          // conv phase: channel within 16-ch slab

    // Block-uniform: is the full 14x22 halo in-bounds? (true for ~79% of blocks)
    const bool interior = (h0 >= 3) && (h0 + TH + 3 <= Hh) &&
                          (w0 >= 3) && (w0 + TW + 3 <= Ww);

    // persistent WMMA accumulators: 6 M-tiles of the 96 output channels
    v8f cacc[6];
    #pragma unroll
    for (int m = 0; m < 6; ++m) {
        #pragma unroll
        for (int r = 0; r < 8; ++r) cacc[m][r] = 0.f;
    }

    for (int chunk = 0; chunk < 3; ++chunk) {           // 3 x 32-channel chunks
        for (int sp = 0; sp < 2; ++sp) {                // 2 x 16-channel slabs
            __syncthreads();                            // protect s_in / s_comb reuse
            // ---- Phase 1: stage padded input slab into LDS ----
            {
                const int chbase = chunk * 32 + sp * 16;
                if (interior) {
                    // CDNA5 async global->LDS: no VGPR landing, tracked by ASYNCcnt
                    for (int i = tid; i < 16 * TIN_H * TIN_W; i += 256) {
                        const int cc = i % TIN_W;
                        const int rr = (i / TIN_W) % TIN_H;
                        const int c  = i / (TIN_W * TIN_H);
                        const float* gp = x + (((size_t)bb * Cc + (chbase + c)) * Hh
                                               + (h0 - 3 + rr)) * Ww + (w0 - 3 + cc);
                        const unsigned loff =
                            (unsigned)(((c * TIN_H + rr) * TIN_W + cc) * 4); // s_in @ LDS 0
                        asm volatile("global_load_async_to_lds_b32 %0, %1, off"
                                     :: "v"(loff), "v"(gp) : "memory");
                    }
                    asm volatile("s_wait_asynccnt 0" ::: "memory");
                } else {
                    // border blocks: checked loads with zero padding ('SAME')
                    for (int i = tid; i < 16 * TIN_H * TIN_W; i += 256) {
                        const int cc = i % TIN_W;
                        const int rr = (i / TIN_W) % TIN_H;
                        const int c  = i / (TIN_W * TIN_H);
                        const int gh = h0 - 3 + rr;
                        const int gw = w0 - 3 + cc;
                        float v = 0.f;
                        if (gh >= 0 && gh < Hh && gw >= 0 && gw < Ww)
                            v = x[(((size_t)bb * Cc + (chbase + c)) * Hh + gh) * Ww + gw];
                        s_in[c][rr][cc] = v;
                    }
                }
            }
            __syncthreads();
            // ---- Phase 2: register-blocked depthwise 3x3/5x5/7x7 column conv ----
            {
                const int ch     = chunk * 32 + sp * 16 + crow;  // global channel
                const int clocal = sp * 16 + crow;               // 0..31 within chunk
                float W3[9], W5[25], W7[49];
                #pragma unroll
                for (int i = 0; i < 9;  ++i) W3[i] = w3[ch * 9  + i];
                #pragma unroll
                for (int i = 0; i < 25; ++i) W5[i] = w5[ch * 25 + i];
                #pragma unroll
                for (int i = 0; i < 49; ++i) W7[i] = w7[ch * 49 + i];
                float a3[8], a5[8], a7[8];
                const float vb3 = b3[ch], vb5 = b5[ch], vb7 = b7[ch];
                #pragma unroll
                for (int o = 0; o < 8; ++o) { a3[o] = vb3; a5[o] = vb5; a7[o] = vb7; }

                #pragma unroll
                for (int r = 0; r < TIN_H; ++r) {       // slide over 14 input rows
                    float xv[7];
                    #pragma unroll
                    for (int k = 0; k < 7; ++k) xv[k] = s_in[crow][r][col + k];
                    #pragma unroll
                    for (int ky = 0; ky < 7; ++ky) {
                        const int o = r - ky;
                        if (o >= 0 && o < 8) {
                            #pragma unroll
                            for (int kx = 0; kx < 7; ++kx)
                                a7[o] += W7[ky * 7 + kx] * xv[kx];
                            if (ky >= 1 && ky <= 5) {
                                #pragma unroll
                                for (int kx = 0; kx < 5; ++kx)
                                    a5[o] += W5[(ky - 1) * 5 + kx] * xv[kx + 1];
                            }
                            if (ky >= 2 && ky <= 4) {
                                #pragma unroll
                                for (int kx = 0; kx < 3; ++kx)
                                    a3[o] += W3[(ky - 2) * 3 + kx] * xv[kx + 2];
                            }
                        }
                    }
                }
                #pragma unroll
                for (int o = 0; o < 8; ++o) {
                    const int px = o * TW + col;
                    s_comb[px][clocal]      = (_Float16)a3[o];   // k-rows  0..31 : x3
                    s_comb[px][32 + clocal] = (_Float16)a5[o];   // k-rows 32..63 : x5
                    s_comb[px][64 + clocal] = (_Float16)a7[o];   // k-rows 64..95 : x7
                }
            }
        }
        __syncthreads();
        // ---- Phase 3: 1x1 fusion conv as WMMA GEMM, accumulate over chunks ----
        {
            const int nrow = lane & 15;          // A: M row / B: N col per ISA layout
            const int kb   = (lane >> 4) << 3;   // 0 or 8: K sub-block of this half-wave
            v16h bfrag[3];
            #pragma unroll
            for (int s = 0; s < 3; ++s) {        // 3 K-steps of 32 (x3|x5|x7 blocks)
                const _Float16* bp = &s_comb[wave * 16 + nrow][s * 32 + kb];
                #pragma unroll
                for (int j = 0; j < 8; ++j) { bfrag[s][j] = bp[j]; bfrag[s][8 + j] = bp[16 + j]; }
            }
            #pragma unroll
            for (int m = 0; m < 6; ++m) {
                #pragma unroll
                for (int s = 0; s < 3; ++s) {
                    // A = wf[16m+M][ s*96 + chunk*32 + K ], fp32 -> f16 on the fly
                    const float* ap = wf + ((size_t)(16 * m + nrow)) * 288 + s * 96 + chunk * 32 + kb;
                    v16h afrag;
                    #pragma unroll
                    for (int j = 0; j < 8; ++j) {
                        afrag[j]     = (_Float16)ap[j];
                        afrag[8 + j] = (_Float16)ap[16 + j];
                    }
                    cacc[m] = __builtin_amdgcn_wmma_f32_16x16x32_f16(
                        false, afrag, false, bfrag[s], (short)0, cacc[m], false, false);
                }
            }
        }
    }
    // ---- Epilogue: bias + coalesced fp32 store ----
    {
        const int nrow = lane & 15;
        const int mhi  = (lane >> 4) << 3;   // lanes 16..31 hold M rows 8..15
        const int hh = h0 + wave;
        const int ww = w0 + nrow;
        #pragma unroll
        for (int m = 0; m < 6; ++m) {
            #pragma unroll
            for (int r = 0; r < 8; ++r) {
                const int o = 16 * m + mhi + r;
                out[(((size_t)bb * Cc + o) * Hh + hh) * Ww + ww] = cacc[m][r] + bfin[o];
            }
        }
    }
}

extern "C" void kernel_launch(void* const* d_in, const int* in_sizes, int n_in,
                              void* d_out, int out_size, void* d_ws, size_t ws_size,
                              hipStream_t stream) {
    const float* x  = (const float*)d_in[0];
    const float* w3 = (const float*)d_in[1];
    const float* b3 = (const float*)d_in[2];
    const float* w5 = (const float*)d_in[3];
    const float* b5 = (const float*)d_in[4];
    const float* w7 = (const float*)d_in[5];
    const float* b7 = (const float*)d_in[6];
    const float* wf = (const float*)d_in[7];
    const float* bf = (const float*)d_in[8];
    dim3 grid(Ww / TW, Hh / TH, 16);   // 14 x 28 x 16 = 6272 blocks
    msconv_fused_wmma<<<grid, 256, 0, stream>>>(x, w3, b3, w5, b5, w7, b7, wf, bf,
                                                (float*)d_out);
}
